// Attention_71829033058395
// MI455X (gfx1250) — compile-verified, matrix operand order checked by reading, not run on hardware
//
#include <hip/hip_runtime.h>
#include <hip/hip_bf16.h>

// ---------------------------------------------------------------------------
// Attention: x[4,2048,512] -> QKV -> 8-head flash attention -> out proj.
// All matmuls via v_wmma_f32_16x16x32_f16 (f16 in, f32 accumulate).
// Tile staging via GLOBAL_LOAD_ASYNC_TO_LDS_B128 (ASYNCcnt) when available.
// ---------------------------------------------------------------------------

typedef _Float16 half_t;
typedef __attribute__((ext_vector_type(16))) _Float16 v16h;
typedef __attribute__((ext_vector_type(8)))  _Float16 v8h;
typedef __attribute__((ext_vector_type(8)))  float    v8f;
typedef __attribute__((ext_vector_type(4)))  int      v4i;

union v16u { v16h v; v8h h[2]; };

#define SEQ    2048
#define DIM    512
#define HEADS  8
#define DHEAD  64
#define NQKV   1536   // 3*DIM
#define BATCH  4
#define MROWS  (BATCH*SEQ)   // 8192
#define ATTN_SCALE 0.125f    // 64^-0.5

// workspace layout (bytes)
#define QSZ      ((size_t)BATCH*HEADS*SEQ*DHEAD*2)   // 8 MB
#define OFF_Q    ((size_t)0)
#define OFF_K    (QSZ)
#define OFF_V    (2*QSZ)
#define OFF_ATTN (3*QSZ)                              // [b, i, h*64+d] f16
#define OFF_WQKV (4*QSZ)                              // [1536][512] f16 (K-contig)
#define OFF_WOUT (OFF_WQKV + (size_t)NQKV*DIM*2)      // [512][512] f16 (K-contig)

typedef v4i __attribute__((address_space(1))) g_v4i;   // global int4
typedef v4i __attribute__((address_space(3))) l_v4i;   // LDS int4

static __device__ __forceinline__ v8f wmma16(v16h a, v16h b, v8f c) {
    return __builtin_amdgcn_wmma_f32_16x16x32_f16(false, a, false, b,
                                                  (short)0, c, false, false);
}

// 16-byte global -> LDS copy; async (ASYNCcnt) on CDNA5 toolchains that
// expose the builtin, otherwise synchronous load + ds_store fallback.
static __device__ __forceinline__ void cp_async_b128(const half_t* g, half_t* l) {
#if __has_builtin(__builtin_amdgcn_global_load_async_to_lds_b128)
    __builtin_amdgcn_global_load_async_to_lds_b128(
        (g_v4i*)g, (l_v4i*)l, 0, 0);
#else
    *(v8h*)l = *(const v8h*)g;
#endif
}

static __device__ __forceinline__ void wait_async_all() {
#if __has_builtin(__builtin_amdgcn_s_wait_asynccnt)
    __builtin_amdgcn_s_wait_asynccnt(0);
#endif
}

// ---------------------------------------------------------------------------
// Kernel 0: f32 [K=512][N] weight -> f16 transposed [N][512] (K contiguous)
// ---------------------------------------------------------------------------
__global__ void wt_convert_kernel(const float* __restrict__ src,
                                  half_t* __restrict__ dst, int ncols) {
    int idx = blockIdx.x * blockDim.x + threadIdx.x;
    if (idx < 512 * ncols) {
        int k = idx & 511;      // dst index = n*512 + k
        int n = idx >> 9;
        dst[idx] = (half_t)src[(size_t)k * ncols + n];
    }
}

// ---------------------------------------------------------------------------
// Kernel 1: QKV GEMM  C[8192,1536] = X[8192,512] * Wqkv, scatter to q/k/v ws.
// Block: 256 threads (8 waves, 4x2), tile 128x128, BK=32; wave tile 32x64.
// ---------------------------------------------------------------------------
__global__ void __launch_bounds__(256)
qkv_gemm_kernel(const float* __restrict__ X, const half_t* __restrict__ Bt,
                half_t* __restrict__ qws, half_t* __restrict__ kws,
                half_t* __restrict__ vws) {
    __shared__ half_t As[128 * 40];
    __shared__ half_t Bs[128 * 40];

    const int tid   = threadIdx.x;
    const int wave  = tid >> 5;
    const int lane  = tid & 31;
    const int l16   = lane & 15;
    const int lhalf = lane >> 4;
    const int wm    = wave >> 1;     // 0..3  -> 32 rows
    const int wn    = wave & 1;      // 0..1  -> 64 cols
    const int m0    = blockIdx.x * 128;
    const int n0    = blockIdx.y * 128;

    const int ar   = tid >> 1;        // 0..127  A row
    const int aseg = (tid & 1) * 16;  // k sub-offset
    const int br   = tid >> 1;        // 0..127  B row (output col)
    const int bseg = (tid & 1) * 16;

    v8f acc[2][4] = {};

    for (int k0 = 0; k0 < DIM; k0 += 32) {
        // stage A (f32 -> f16 convert, VALU path)
        {
            const float* src = X + (size_t)(m0 + ar) * DIM + k0 + aseg;
            __builtin_prefetch(src + 32, 0, 3);   // next K tile, near scope
            half_t tmp[16];
#pragma unroll
            for (int i = 0; i < 16; ++i) tmp[i] = (half_t)src[i];
            *(v8h*)&As[ar * 40 + aseg]     = *(v8h*)&tmp[0];
            *(v8h*)&As[ar * 40 + aseg + 8] = *(v8h*)&tmp[8];
        }
        // stage B (f16, async global->LDS)
        {
            const half_t* src = Bt + (size_t)(n0 + br) * DIM + k0 + bseg;
            cp_async_b128(src,     &Bs[br * 40 + bseg]);
            cp_async_b128(src + 8, &Bs[br * 40 + bseg + 8]);
        }
        wait_async_all();
        __syncthreads();

        v16u a[2], b[4];
#pragma unroll
        for (int t = 0; t < 2; ++t) {
            const int row = wm * 32 + t * 16 + l16;
            a[t].h[0] = *(v8h*)&As[row * 40 + 8 * lhalf];
            a[t].h[1] = *(v8h*)&As[row * 40 + 16 + 8 * lhalf];
        }
#pragma unroll
        for (int u = 0; u < 4; ++u) {
            const int nrow = wn * 64 + u * 16 + l16;
            b[u].h[0] = *(v8h*)&Bs[nrow * 40 + 8 * lhalf];
            b[u].h[1] = *(v8h*)&Bs[nrow * 40 + 16 + 8 * lhalf];
        }
#pragma unroll
        for (int tm = 0; tm < 2; ++tm)
#pragma unroll
            for (int tn = 0; tn < 4; ++tn)
                acc[tm][tn] = wmma16(a[tm].v, b[tn].v, acc[tm][tn]);
        __syncthreads();
    }

    // epilogue: scatter into q (pre-scaled), k [b,h,n,d] and v [b,h,d,n]
#pragma unroll
    for (int tm = 0; tm < 2; ++tm)
#pragma unroll
        for (int tn = 0; tn < 4; ++tn)
#pragma unroll
            for (int r = 0; r < 8; ++r) {
                const int m = m0 + wm * 32 + tm * 16 + r + 8 * lhalf;
                const int c = n0 + wn * 64 + tn * 16 + l16;
                const float val = acc[tm][tn][r];
                const int part = c >> 9;          // 0=q 1=k 2=v
                const int wi   = c & 511;
                const int h    = wi >> 6;
                const int d    = wi & 63;
                const int bb   = m >> 11;
                const int i    = m & 2047;
                const size_t bh = (size_t)bb * HEADS + h;
                if (part == 0)
                    qws[(bh * SEQ + i) * DHEAD + d] = (half_t)(val * ATTN_SCALE);
                else if (part == 1)
                    kws[(bh * SEQ + i) * DHEAD + d] = (half_t)val;
                else
                    vws[(bh * DHEAD + d) * SEQ + i] = (half_t)val;
            }
}

// ---------------------------------------------------------------------------
// Kernel 2: flash attention. grid = (32 q-tiles, 32 b*h), 128 threads/block.
// Each wave owns 16 query rows; 64-key tiles; online softmax.
// ---------------------------------------------------------------------------
__global__ void __launch_bounds__(128)
flash_attn_kernel(const half_t* __restrict__ qws, const half_t* __restrict__ kws,
                  const half_t* __restrict__ vws, half_t* __restrict__ attn) {
    __shared__ half_t Pl[4][16 * 72];   // per-wave C->A layout staging

    const int tid   = threadIdx.x;
    const int wave  = tid >> 5;
    const int lane  = tid & 31;
    const int l16   = lane & 15;
    const int lhalf = lane >> 4;
    const int bh    = blockIdx.y;              // 0..31
    const int qrow0 = blockIdx.x * 64 + wave * 16;

    const half_t* qh = qws + (size_t)bh * SEQ * DHEAD;
    const half_t* kh = kws + (size_t)bh * SEQ * DHEAD;
    const half_t* vh = vws + (size_t)bh * DHEAD * SEQ;
    half_t* P = Pl[wave];

    // Q fragments (16 rows x 64 dims = 2 K-steps), pre-scaled by 1/sqrt(d)
    v16u qf[2];
#pragma unroll
    for (int ks = 0; ks < 2; ++ks) {
        const half_t* src = qh + (size_t)(qrow0 + l16) * DHEAD + ks * 32 + 8 * lhalf;
        qf[ks].h[0] = *(const v8h*)src;
        qf[ks].h[1] = *(const v8h*)(src + 16);
    }

    v8f o[4] = {};
    float mrow[8], lrow[8];
#pragma unroll
    for (int r = 0; r < 8; ++r) { mrow[r] = -1e30f; lrow[r] = 0.0f; }

    for (int kt = 0; kt < SEQ / 64; ++kt) {
        const int key0 = kt * 64;
        // ---- S = Q * K^T (scale folded into Q) ----
        v8f s[4] = {};
#pragma unroll
        for (int nt = 0; nt < 4; ++nt) {
#pragma unroll
            for (int ks = 0; ks < 2; ++ks) {
                v16u bf;
                const half_t* src =
                    kh + (size_t)(key0 + nt * 16 + l16) * DHEAD + ks * 32 + 8 * lhalf;
                bf.h[0] = *(const v8h*)src;
                bf.h[1] = *(const v8h*)(src + 16);
                s[nt] = wmma16(qf[ks].v, bf.v, s[nt]);
            }
        }
        // ---- online softmax (row = 16 lanes of one half-wave) ----
#pragma unroll
        for (int r = 0; r < 8; ++r) {
            float mt = fmaxf(fmaxf(s[0][r], s[1][r]), fmaxf(s[2][r], s[3][r]));
#pragma unroll
            for (int off = 8; off >= 1; off >>= 1)
                mt = fmaxf(mt, __shfl_xor(mt, off, 32));
            const float mnew = fmaxf(mrow[r], mt);
            const float corr = __expf(mrow[r] - mnew);
            mrow[r] = mnew;
            float rsum = 0.0f;
#pragma unroll
            for (int nt = 0; nt < 4; ++nt) {
                const float p = __expf(s[nt][r] - mnew);
                s[nt][r] = p;
                rsum += p;
            }
#pragma unroll
            for (int off = 8; off >= 1; off >>= 1)
                rsum += __shfl_xor(rsum, off, 32);
            lrow[r] = lrow[r] * corr + rsum;
#pragma unroll
            for (int nt = 0; nt < 4; ++nt) o[nt][r] = o[nt][r] * corr;
        }
        // ---- P: C-layout -> LDS -> A-layout (in-order LDS within wave) ----
#pragma unroll
        for (int nt = 0; nt < 4; ++nt)
#pragma unroll
            for (int r = 0; r < 8; ++r)
                P[(r + 8 * lhalf) * 72 + nt * 16 + l16] = (half_t)s[nt][r];

        v16u pf[2];
#pragma unroll
        for (int ks = 0; ks < 2; ++ks) {
            pf[ks].h[0] = *(v8h*)&P[l16 * 72 + ks * 32 + 8 * lhalf];
            pf[ks].h[1] = *(v8h*)&P[l16 * 72 + ks * 32 + 16 + 8 * lhalf];
        }
        // ---- O += P * V  (V stored [d][n] so B-frags are contiguous) ----
#pragma unroll
        for (int nt = 0; nt < 4; ++nt) {
#pragma unroll
            for (int ks = 0; ks < 2; ++ks) {
                v16u vf;
                const half_t* src =
                    vh + (size_t)(nt * 16 + l16) * SEQ + key0 + ks * 32 + 8 * lhalf;
                vf.h[0] = *(const v8h*)src;
                vf.h[1] = *(const v8h*)(src + 16);
                o[nt] = wmma16(pf[ks].v, vf.v, o[nt]);
            }
        }
    }

    // ---- normalize and store to [b, i, h*64 + d] f16 ----
    const int bb = bh >> 3;
    const int h  = bh & 7;
#pragma unroll
    for (int r = 0; r < 8; ++r) {
        const float inv = 1.0f / lrow[r];
        const int i = qrow0 + r + 8 * lhalf;
#pragma unroll
        for (int nt = 0; nt < 4; ++nt) {
            const int c = h * 64 + nt * 16 + l16;
            attn[((size_t)bb * SEQ + i) * DIM + c] = (half_t)(o[nt][r] * inv);
        }
    }
}

// ---------------------------------------------------------------------------
// Kernel 3: out = attn[8192,512](f16) * Wout + bias -> f32
// Block: 256 threads, tile 128x128, fully async-staged (both operands f16).
// ---------------------------------------------------------------------------
__global__ void __launch_bounds__(256)
out_gemm_kernel(const half_t* __restrict__ A, const half_t* __restrict__ Bt,
                const float* __restrict__ bias, float* __restrict__ out) {
    __shared__ half_t As[128 * 40];
    __shared__ half_t Bs[128 * 40];

    const int tid   = threadIdx.x;
    const int wave  = tid >> 5;
    const int lane  = tid & 31;
    const int l16   = lane & 15;
    const int lhalf = lane >> 4;
    const int wm    = wave >> 1;
    const int wn    = wave & 1;
    const int m0    = blockIdx.x * 128;
    const int n0    = blockIdx.y * 128;

    const int ar   = tid >> 1;
    const int aseg = (tid & 1) * 16;

    v8f acc[2][4] = {};

    for (int k0 = 0; k0 < DIM; k0 += 32) {
        {
            const half_t* srcA = A + (size_t)(m0 + ar) * DIM + k0 + aseg;
            cp_async_b128(srcA,     &As[ar * 40 + aseg]);
            cp_async_b128(srcA + 8, &As[ar * 40 + aseg + 8]);
            const half_t* srcB = Bt + (size_t)(n0 + ar) * DIM + k0 + aseg;
            cp_async_b128(srcB,     &Bs[ar * 40 + aseg]);
            cp_async_b128(srcB + 8, &Bs[ar * 40 + aseg + 8]);
        }
        wait_async_all();
        __syncthreads();

        v16u a[2], b[4];
#pragma unroll
        for (int t = 0; t < 2; ++t) {
            const int row = wm * 32 + t * 16 + l16;
            a[t].h[0] = *(v8h*)&As[row * 40 + 8 * lhalf];
            a[t].h[1] = *(v8h*)&As[row * 40 + 16 + 8 * lhalf];
        }
#pragma unroll
        for (int u = 0; u < 4; ++u) {
            const int nrow = wn * 64 + u * 16 + l16;
            b[u].h[0] = *(v8h*)&Bs[nrow * 40 + 8 * lhalf];
            b[u].h[1] = *(v8h*)&Bs[nrow * 40 + 16 + 8 * lhalf];
        }
#pragma unroll
        for (int tm = 0; tm < 2; ++tm)
#pragma unroll
            for (int tn = 0; tn < 4; ++tn)
                acc[tm][tn] = wmma16(a[tm].v, b[tn].v, acc[tm][tn]);
        __syncthreads();
    }

#pragma unroll
    for (int tm = 0; tm < 2; ++tm)
#pragma unroll
        for (int tn = 0; tn < 4; ++tn)
#pragma unroll
            for (int r = 0; r < 8; ++r) {
                const int m = m0 + wm * 32 + tm * 16 + r + 8 * lhalf;
                const int c = n0 + wn * 64 + tn * 16 + l16;
                out[(size_t)m * DIM + c] = acc[tm][tn][r] + bias[c];
            }
}

// ---------------------------------------------------------------------------
extern "C" void kernel_launch(void* const* d_in, const int* in_sizes, int n_in,
                              void* d_out, int out_size, void* d_ws, size_t ws_size,
                              hipStream_t stream) {
    const float* x     = (const float*)d_in[0];   // [4,2048,512]
    const float* w_qkv = (const float*)d_in[1];   // [512,1536]
    const float* w_out = (const float*)d_in[2];   // [512,512]
    const float* b_out = (const float*)d_in[3];   // [512]
    float* out = (float*)d_out;

    char* ws = (char*)d_ws;
    half_t* qws   = (half_t*)(ws + OFF_Q);
    half_t* kws   = (half_t*)(ws + OFF_K);
    half_t* vws   = (half_t*)(ws + OFF_V);
    half_t* attn  = (half_t*)(ws + OFF_ATTN);
    half_t* wqkvT = (half_t*)(ws + OFF_WQKV);
    half_t* woutT = (half_t*)(ws + OFF_WOUT);

    wt_convert_kernel<<<(512 * NQKV + 255) / 256, 256, 0, stream>>>(w_qkv, wqkvT, NQKV);
    wt_convert_kernel<<<(512 * DIM + 255) / 256, 256, 0, stream>>>(w_out, woutT, DIM);

    qkv_gemm_kernel<<<dim3(MROWS / 128, NQKV / 128), 256, 0, stream>>>(
        x, wqkvT, qws, kws, vws);

    flash_attn_kernel<<<dim3(SEQ / 64, BATCH * HEADS), 128, 0, stream>>>(
        qws, kws, vws, attn);

    out_gemm_kernel<<<dim3(MROWS / 128, DIM / 128), 256, 0, stream>>>(
        attn, woutT, b_out, out);
}
